// GiPHEmbedding_GraphSAGE_49701361549771
// MI455X (gfx1250) — matrix operand order repaired and checked by reading, not compiled, not run on hardware
//
#include <hip/hip_runtime.h>
#include <hip/hip_bf16.h>

#define NNODES 50000
#define DIM    128
#define MTILES (NNODES / 16)   // 3125, exact

typedef __attribute__((ext_vector_type(16))) __bf16 v16bf;
typedef __attribute__((ext_vector_type(8)))  float  v8f;

// ---------------- utility kernels ----------------

__global__ void zero_f32(float* __restrict__ p, int n) {
    int i = blockIdx.x * blockDim.x + threadIdx.x;
    if (i < n) p[i] = 0.0f;
}

__global__ void degree_kernel(const int* __restrict__ dst, float* __restrict__ deg, int ne) {
    int e = blockIdx.x * blockDim.x + threadIdx.x;
    if (e < ne) atomicAdd(&deg[dst[e]], 1.0f);
}

__global__ void invdeg_kernel(const float* __restrict__ deg, float* __restrict__ invdeg, int n) {
    int i = blockIdx.x * blockDim.x + threadIdx.x;
    if (i < n) invdeg[i] = 1.0f / fmaxf(deg[i], 1.0f);
}

// ---------------- edge scatter (segment_sum) ----------------
// One wave32 per edge: lane loads float4 of h[src], atomically adds into acc[dst].
// Node tables are L2-resident (25.6 MB << 192 MB L2): this is L2-atomic bound.
__global__ __launch_bounds__(256) void scatter_add(
    const float* __restrict__ h, const int* __restrict__ src, const int* __restrict__ dst,
    float* __restrict__ acc, int ne)
{
    int gid  = blockIdx.x * blockDim.x + threadIdx.x;
    int e    = gid >> 5;
    int lane = gid & 31;
    if (e >= ne) return;
    int s = src[e];
    int d = dst[e];
    const float4 v = *(const float4*)(h + (size_t)s * DIM + lane * 4);
    float* p = acc + (size_t)d * DIM + lane * 4;
    atomicAdd(p + 0, v.x);
    atomicAdd(p + 1, v.y);
    atomicAdd(p + 2, v.z);
    atomicAdd(p + 3, v.w);
}

// ---------------- fused SAGE layer GEMM ----------------
// out[row, :] = relu?( h[row,:] @ Wself^T + b + (acc[row,:]*invdeg[row]) @ Wneigh^T )
// bf16 WMMA, f32 accumulate, activations split hi/lo for ~fp32 accuracy.

__device__ __forceinline__ void cvt_split8(const float* p, float scale,
                                           v16bf& hi, v16bf& lo, int dstoff) {
    float4 a = *(const float4*)(p);
    float4 b = *(const float4*)(p + 4);
    float f[8] = {a.x, a.y, a.z, a.w, b.x, b.y, b.z, b.w};
#pragma unroll
    for (int j = 0; j < 8; ++j) {
        float v  = f[j] * scale;
        __bf16 h = (__bf16)v;
        hi[dstoff + j] = h;
        lo[dstoff + j] = (__bf16)(v - (float)h);
    }
}

__global__ __launch_bounds__(256) void sage_gemm(
    const float* __restrict__ h,       // [N,128]
    const float* __restrict__ acc,     // [N,128] neighbor sums
    const float* __restrict__ invdeg,  // [N]
    const float* __restrict__ Wself,   // [128,128] row-major [out][in]
    const float* __restrict__ Wneigh,  // [128,128]
    const float* __restrict__ bias,    // [128]
    float* __restrict__ out,           // [N,128]
    int do_relu)
{
    __shared__ __attribute__((aligned(32))) __bf16 sWs[DIM * DIM]; // 32 KB
    __shared__ __attribute__((aligned(32))) __bf16 sWn[DIM * DIM]; // 32 KB

    int t = threadIdx.x;
    // cooperative f32 -> bf16 weight stage into LDS (64 KB, reused by 8 waves)
    for (int i = t; i < DIM * DIM; i += 256) {
        sWs[i] = (__bf16)Wself[i];
        sWn[i] = (__bf16)Wneigh[i];
    }
    __syncthreads();

    int wave  = t >> 5;
    int lane  = t & 31;
    int mtile = blockIdx.x * 8 + wave;
    if (mtile >= MTILES) return;

    int m   = lane & 15;   // A row within tile / B column within n-tile
    int hi2 = lane >> 4;   // lane-half selector
    int row = mtile * 16 + m;
    float idg = invdeg[row];

    const float* hrow = h   + (size_t)row * DIM;
    const float* arow = acc + (size_t)row * DIM;

    v8f c[8];
#pragma unroll
    for (int i = 0; i < 8; ++i) c[i] = {};

#pragma unroll
    for (int kt = 0; kt < 4; ++kt) {
        int k0   = kt * 32;
        int base = k0 + hi2 * 16;  // A layout: lanes 0-15 -> K 0-15 ; lanes 16-31 -> K 16-31

        v16bf As_hi, As_lo, An_hi, An_lo;
        cvt_split8(hrow + base,      1.0f, As_hi, As_lo, 0);
        cvt_split8(hrow + base + 8,  1.0f, As_hi, As_lo, 8);
        cvt_split8(arow + base,      idg,  An_hi, An_lo, 0);
        cvt_split8(arow + base + 8,  idg,  An_hi, An_lo, 8);

        int koff = k0 + hi2 * 16;  // B layout: 16 consecutive K per lane
#pragma unroll
        for (int nt = 0; nt < 8; ++nt) {
            int n = nt * 16 + m;
            v16bf Bs = *(const v16bf*)(sWs + n * DIM + koff);
            v16bf Bn = *(const v16bf*)(sWn + n * DIM + koff);
            c[nt] = __builtin_amdgcn_wmma_f32_16x16x32_bf16(false, As_hi, false, Bs, (short)0, c[nt], false, false);
            c[nt] = __builtin_amdgcn_wmma_f32_16x16x32_bf16(false, As_lo, false, Bs, (short)0, c[nt], false, false);
            c[nt] = __builtin_amdgcn_wmma_f32_16x16x32_bf16(false, An_hi, false, Bn, (short)0, c[nt], false, false);
            c[nt] = __builtin_amdgcn_wmma_f32_16x16x32_bf16(false, An_lo, false, Bn, (short)0, c[nt], false, false);
        }
    }

    // epilogue: C/D layout — VGPR r: M = r + 8*hi2, N = nt*16 + m
#pragma unroll
    for (int nt = 0; nt < 8; ++nt) {
        int n   = nt * 16 + m;
        float b = bias[n];
#pragma unroll
        for (int r = 0; r < 8; ++r) {
            int orow = mtile * 16 + r + hi2 * 8;
            float v  = c[nt][r] + b;
            if (do_relu) v = fmaxf(v, 0.0f);
            out[(size_t)orow * DIM + n] = v;
        }
    }
}

// ---------------- host-side orchestration ----------------

extern "C" void kernel_launch(void* const* d_in, const int* in_sizes, int n_in,
                              void* d_out, int out_size, void* d_ws, size_t ws_size,
                              hipStream_t stream) {
    const float* x   = (const float*)d_in[0];
    const int*   src = (const int*)d_in[1];
    const int*   dst = (const int*)d_in[2];
    const float* Wn1 = (const float*)d_in[3];
    const float* Ws1 = (const float*)d_in[4];
    const float* b1  = (const float*)d_in[5];
    const float* Wn2 = (const float*)d_in[6];
    const float* Ws2 = (const float*)d_in[7];
    const float* b2  = (const float*)d_in[8];
    const float* Wn3 = (const float*)d_in[9];
    const float* Ws3 = (const float*)d_in[10];
    const float* b3  = (const float*)d_in[11];
    float* out = (float*)d_out;

    const int ne = in_sizes[1];
    const int nf = NNODES * DIM;

    float* ws     = (float*)d_ws;
    float* deg    = ws;                 // N
    float* invdeg = ws + NNODES;        // N
    float* accb   = ws + 2 * NNODES;    // N*128
    float* h1     = accb + nf;          // N*128
    float* h2     = h1 + nf;            // N*128

    const int B = 256;
    int gNodes = (NNODES + B - 1) / B;
    int gFeat  = (nf + B - 1) / B;
    int gEdge  = (ne + B - 1) / B;
    int gScat  = (int)(((long long)ne * 32 + B - 1) / B);
    int gGemm  = (MTILES + 7) / 8;

    // degrees (shared by all three layers, recomputed each call for determinism)
    zero_f32<<<gNodes, B, 0, stream>>>(deg, NNODES);
    degree_kernel<<<gEdge, B, 0, stream>>>(dst, deg, ne);
    invdeg_kernel<<<gNodes, B, 0, stream>>>(deg, invdeg, NNODES);

    // layer 1: x -> h1 (relu)
    zero_f32<<<gFeat, B, 0, stream>>>(accb, nf);
    scatter_add<<<gScat, B, 0, stream>>>(x, src, dst, accb, ne);
    sage_gemm<<<gGemm, B, 0, stream>>>(x, accb, invdeg, Ws1, Wn1, b1, h1, 1);

    // layer 2: h1 -> h2 (relu)
    zero_f32<<<gFeat, B, 0, stream>>>(accb, nf);
    scatter_add<<<gScat, B, 0, stream>>>(h1, src, dst, accb, ne);
    sage_gemm<<<gGemm, B, 0, stream>>>(h1, accb, invdeg, Ws2, Wn2, b2, h2, 1);

    // layer 3: h2 -> out (no relu)
    zero_f32<<<gFeat, B, 0, stream>>>(accb, nf);
    scatter_add<<<gScat, B, 0, stream>>>(h2, src, dst, accb, ne);
    sage_gemm<<<gGemm, B, 0, stream>>>(h2, accb, invdeg, Ws3, Wn3, b3, out, 0);
}
